// LSTM_3332894622294
// MI455X (gfx1250) — matrix operand order repaired and checked by reading, not compiled, not run on hardware
//
#include <hip/hip_runtime.h>
#include <hip/hip_bf16.h>

typedef __bf16 bf16_t;
typedef bf16_t v16bf __attribute__((ext_vector_type(16)));
typedef bf16_t v8bf  __attribute__((ext_vector_type(8)));
typedef float  v8f   __attribute__((ext_vector_type(8)));
typedef int    v4i   __attribute__((ext_vector_type(4)));

#define T_STEPS 2048
#define BATCH   32
#define DIN     256
#define DH      256
#define NWG     16           // persistent workgroups
#define UPG     16           // hidden units per WG
#define NT      4            // 16x16 N tiles per WG
#define KT      16           // K tiles of 32 (8 x-part + 8 h-part)
#define PSTR    66           // padded float stride of preact in LDS
#define XSTEP   8192         // halfwords per step of packed x: 2 mtiles * 8 kt * 32 lanes * 16 e

// ---- workspace layout (halfword units) ----
#define XT_HALF  ((size_t)T_STEPS * XSTEP)
#define WP_HALF  ((size_t)NWG * NT * KT * 32 * 16)
#define HB_HALF  ((size_t)2 * BATCH * DH)

#if __has_builtin(__builtin_amdgcn_global_load_async_to_lds_b128)
#define USE_ASYNC_X 1
// builtin signature (from hipcc diagnostic): (int4 AS1* src, int4 AS3* dst, imm offset, imm cpol)
typedef __attribute__((address_space(1))) v4i as1_v4i;
typedef __attribute__((address_space(3))) v4i as3_v4i;
#else
#define USE_ASYNC_X 0
#endif

#if __has_builtin(__builtin_amdgcn_sched_barrier)
#define SCHED_FENCE() __builtin_amdgcn_sched_barrier(0)
#else
#define SCHED_FENCE() asm volatile("" ::: "memory")
#endif

// ---------------- preprocessing kernels ----------------

__global__ __launch_bounds__(256) void cvt_x_kernel(const float* __restrict__ x,
                                                    bf16_t* __restrict__ xF) {
    // Pack x into WMMA A-fragment order. 16,777,216 elements.
    unsigned idx  = blockIdx.x * 256u + threadIdx.x;
    unsigned e    = idx & 15u;
    unsigned lane = (idx >> 4) & 31u;
    unsigned kt   = (idx >> 9) & 7u;
    unsigned mt   = (idx >> 12) & 1u;
    unsigned t    = idx >> 13;
    unsigned m    = mt * 16 + (lane & 15u);
    unsigned K    = kt * 32 + ((lane < 16u) ? 0u : 8u) + ((e < 8u) ? e : (e + 8u));
    xF[idx] = (bf16_t)x[((size_t)m * T_STEPS + t) * DIN + K];
}

__global__ __launch_bounds__(256) void pack_w_kernel(const float* __restrict__ W,
                                                     const float* __restrict__ U,
                                                     bf16_t* __restrict__ Wp) {
    // Fragment-native pack of [W;U] (512 x 1024) into per-WG slices. 524,288 elements.
    unsigned idx  = blockIdx.x * 256u + threadIdx.x;
    unsigned e    = idx & 15u;
    unsigned lane = (idx >> 4) & 31u;
    unsigned kt   = (idx >> 9) & 15u;
    unsigned nt   = (idx >> 13) & 3u;
    unsigned g    = idx >> 15;
    unsigned nloc = nt * 16 + (lane & 15u);            // [0,64) within WG
    unsigned gate = nloc >> 4;                         // 0=i 1=f 2=o 3=g
    unsigned u    = nloc & 15u;
    unsigned col  = gate * DH + g * UPG + u;           // column of [W;U] (4D = 1024)
    unsigned K    = kt * 32 + ((lane < 16u) ? 0u : 8u) + ((e < 8u) ? e : (e + 8u));
    float v = (K < DIN) ? W[(size_t)K * (4 * DH) + col]
                        : U[(size_t)(K - DIN) * (4 * DH) + col];
    Wp[idx] = (bf16_t)v;
}

__global__ __launch_bounds__(256) void init_state_kernel(bf16_t* __restrict__ hb,
                                                         unsigned* __restrict__ ctr) {
    unsigned idx = blockIdx.x * 256u + threadIdx.x;
    if (idx < (unsigned)HB_HALF) hb[idx] = (bf16_t)0.0f;
    if (idx == (unsigned)HB_HALF) *ctr = 0u;
}

// ---------------- persistent LSTM scan kernel ----------------

__global__ __launch_bounds__(256) void lstm_scan_kernel(const bf16_t* __restrict__ xF,
                                                        const bf16_t* __restrict__ Wp,
                                                        const float* __restrict__ bias,
                                                        bf16_t* __restrict__ hb,
                                                        unsigned* __restrict__ ctr,
                                                        float* __restrict__ out) {
    __shared__ __align__(16) float Pre[BATCH * PSTR];   // preact slice (32 x 64)
    __shared__ float               Cst[BATCH * UPG];    // cell state slice (32 x 16)
#if USE_ASYNC_X
    __shared__ __align__(16) bf16_t Xb[2][XSTEP];       // async x-fragment ping-pong (32 KB)
#endif

    const int g    = blockIdx.x;
    const int tid  = threadIdx.x;
    const int lane = tid & 31;
    const int wave = tid >> 5;
    const int mtile = wave & 1;      // 2 M tiles (batch 0-15 / 16-31)
    const int ntile = wave >> 1;     // 4 N tiles -> 8 waves == 8 C tiles

    for (int i = tid; i < BATCH * UPG; i += 256) Cst[i] = 0.0f;

    const bf16_t* wgW   = Wp + (size_t)g * (NT * KT * 32 * 16);
    const bf16_t* bbase = wgW + ((size_t)(ntile * KT) * 32 + lane) * 16;  // 32B/lane per kt
    const int hrow = mtile * 16 + (lane & 15);
    const int hsel = (lane < 16) ? 0 : 8;               // h-frag K start within tile
    const int xoff = ((mtile * 8) * 32 + lane) * 16;    // this wave's x-frag base (halfwords)

    union frag { v16bf v; v8bf h[2]; };

#if USE_ASYNC_X
    // preload x fragments of step 0 into Xb[0]
    for (int i = 0; i < 4; ++i) {
        int c = tid + i * 256;                          // 1024 chunks of 16B
        __builtin_amdgcn_global_load_async_to_lds_b128(
            (as1_v4i*)(xF + (size_t)c * 8),
            (as3_v4i*)(&Xb[0][c * 8]), 0, 0);
    }
#endif

    for (int t = 0; t < T_STEPS; ++t) {
        // ---- wait for all WGs to have published h for step t ----
        __builtin_amdgcn_s_cluster_barrier();           // NOP when not cluster-launched
        if (tid == 0) {
            const unsigned want = (unsigned)(NWG * t);
            while (__hip_atomic_load(ctr, __ATOMIC_ACQUIRE, __HIP_MEMORY_SCOPE_AGENT) < want)
                __builtin_amdgcn_s_sleep(2);
        }
#if USE_ASYNC_X
        __builtin_amdgcn_s_wait_asynccnt(0);            // step-t x frags resident in LDS
#endif
        __syncthreads();

        const int cur = t & 1;
        const bf16_t* hsrc = hb + (size_t)cur * (BATCH * DH);

#if USE_ASYNC_X
        // ---- async prefetch of step t+1 x fragments into the other LDS buffer ----
        if (t + 1 < T_STEPS) {
            const bf16_t* gsrc = xF + (size_t)(t + 1) * XSTEP;
            bf16_t* ldst = &Xb[(t + 1) & 1][0];
            #pragma unroll
            for (int i = 0; i < 4; ++i) {
                int c = tid + i * 256;
                __builtin_amdgcn_global_load_async_to_lds_b128(
                    (as1_v4i*)(gsrc + (size_t)c * 8),
                    (as3_v4i*)(ldst + c * 8), 0, 0);
            }
        }
        const bf16_t* xbase = &Xb[cur][xoff];           // LDS reads (ds_load_b128)
#else
        const bf16_t* xbase = xF + (size_t)t * XSTEP + xoff;   // direct global reads
        if (t + 1 < T_STEPS && tid < 64)
            __builtin_prefetch(xF + (size_t)(t + 1) * XSTEP + tid * 128, 0, 1);
#endif

        // ---- preload ALL A fragments into distinct registers, keep every load in
        //      flight before the first WMMA (sched fence stops the scheduler from
        //      collapsing them back into load->wait(0)->wmma lockstep) ----
        frag ah[8];
        #pragma unroll
        for (int kt = 0; kt < 8; ++kt) {                // h-part first: longest latency (L2)
            const bf16_t* hp = hsrc + hrow * DH + kt * 32 + hsel;
            ah[kt].h[0] = *(const v8bf*)(hp);
            ah[kt].h[1] = *(const v8bf*)(hp + 16);
        }
        frag ax[8];
        #pragma unroll
        for (int kt = 0; kt < 8; ++kt) {                // x-part: short latency (LDS/L2)
            const bf16_t* xp = xbase + kt * (32 * 16);
            ax[kt].h[0] = *(const v8bf*)(xp);
            ax[kt].h[1] = *(const v8bf*)(xp + 8);
        }
        SCHED_FENCE();

        // ---- two independent WMMA chains: x-chain runs while h loads land ----
        v8f acc0 = {}, acc1 = {};
        #pragma unroll
        for (int kt = 0; kt < 8; ++kt) {
            union { v16bf v; v8bf h[2]; } b;
            const bf16_t* bp = bbase + (size_t)kt * (32 * 16);
            b.h[0] = *(const v8bf*)(bp);
            b.h[1] = *(const v8bf*)(bp + 8);
            acc0 = __builtin_amdgcn_wmma_f32_16x16x32_bf16(
                       false, ax[kt].v, false, b.v, (short)0, acc0, false, false);
        }
        #pragma unroll
        for (int kt = 0; kt < 8; ++kt) {
            union { v16bf v; v8bf h[2]; } b;
            const bf16_t* bp = bbase + (size_t)(8 + kt) * (32 * 16);
            b.h[0] = *(const v8bf*)(bp);
            b.h[1] = *(const v8bf*)(bp + 8);
            acc1 = __builtin_amdgcn_wmma_f32_16x16x32_bf16(
                       false, ah[kt].v, false, b.v, (short)0, acc1, false, false);
        }

        {   // C layout: VGPR r, lanes0-15 -> M=r, lanes16-31 -> M=r+8, N=lane%16
            const int m0 = mtile * 16 + ((lane < 16) ? 0 : 8);
            const int n  = ntile * 16 + (lane & 15);
            #pragma unroll
            for (int r = 0; r < 8; ++r) Pre[(m0 + r) * PSTR + n] = acc0[r] + acc1[r];
        }
        __syncthreads();

        // ---- gates + state update for this WG's 16 hidden units ----
        bf16_t* hdst = hb + (size_t)(1 - cur) * (BATCH * DH);
        for (int p = tid; p < BATCH * UPG; p += 256) {
            const int m = p >> 4, u = p & 15;
            const int gu = g * UPG + u;                 // global hidden unit
            const float pi = Pre[m * PSTR + u]        + bias[0 * DH + gu];
            const float pf = Pre[m * PSTR + 16 + u]   + bias[1 * DH + gu];
            const float po = Pre[m * PSTR + 32 + u]   + bias[2 * DH + gu];
            const float pg = Pre[m * PSTR + 48 + u]   + bias[3 * DH + gu];
            const float iv = 1.0f / (1.0f + __expf(-pi));
            const float fv = 1.0f / (1.0f + __expf(-pf));
            const float ov = 1.0f / (1.0f + __expf(-po));
            const float gv = 2.0f / (1.0f + __expf(-2.0f * pg)) - 1.0f;   // tanh
            const float cn = fv * Cst[p] + iv * gv;
            Cst[p] = cn;
            const float th = 2.0f / (1.0f + __expf(-2.0f * cn)) - 1.0f;
            const float hv = ov * th;
            out[((size_t)m * T_STEPS + t) * DH + gu] = hv;
            hdst[m * DH + gu] = (bf16_t)hv;
        }
        __syncthreads();
        // ---- publish: release h for step t+1 ----
        if (tid == 0)
            __hip_atomic_fetch_add(ctr, 1u, __ATOMIC_RELEASE, __HIP_MEMORY_SCOPE_AGENT);
    }
}

// ---------------- launcher ----------------

extern "C" void kernel_launch(void* const* d_in, const int* in_sizes, int n_in,
                              void* d_out, int out_size, void* d_ws, size_t ws_size,
                              hipStream_t stream) {
    const float* x = (const float*)d_in[0];   // [32, 2048, 256]
    const float* W = (const float*)d_in[1];   // [256, 1024]
    const float* U = (const float*)d_in[2];   // [256, 1024]
    const float* b = (const float*)d_in[3];   // [1024]
    float* out = (float*)d_out;               // [32, 2048, 256]

    bf16_t* xF  = (bf16_t*)d_ws;
    bf16_t* Wp  = xF + XT_HALF;
    bf16_t* hbf = Wp + WP_HALF;
    unsigned* ctr = (unsigned*)(hbf + HB_HALF);

    cvt_x_kernel<<<(unsigned)(XT_HALF / 256), 256, 0, stream>>>(x, xF);
    pack_w_kernel<<<(unsigned)(WP_HALF / 256), 256, 0, stream>>>(W, U, Wp);
    init_state_kernel<<<(unsigned)((HB_HALF + 256) / 256 + 1), 256, 0, stream>>>(hbf, ctr);
    lstm_scan_kernel<<<NWG, 256, 0, stream>>>(xF, Wp, b, hbf, ctr, out);
}
